// ElemExtractor_6828998001219
// MI455X (gfx1250) — matrix operand order repaired and checked by reading, not compiled, not run on hardware
//
#include <hip/hip_runtime.h>

typedef __attribute__((ext_vector_type(16))) _Float16 v16h;
typedef __attribute__((ext_vector_type(8)))  _Float16 v8h;
typedef __attribute__((ext_vector_type(8)))  float    v8f;

#define Bq 64
#define Lq 512
#define Dq 768
#define Sq 64
#define F1 384
#define F2 128
#define M_TOTAL (Bq * Sq)   // 4096 rows

#define XS_STRIDE 776       // halfs per row (97 * 8) -> bank-conflict-free
#define H1_STRIDE 392       // 384 + 8
#define H2_STRIDE 136       // 128 + 8

// ---------------- prep kernels ----------------

__global__ void k_zero16(uint4* __restrict__ p, int n) {
  int i = blockIdx.x * blockDim.x + threadIdx.x;
  if (i < n) p[i] = uint4{0u, 0u, 0u, 0u};
}

// w: [K][N] row-major f32  ->  wt: [N][K] f16 (transposed, coalesced writes)
__global__ void k_cvt_t(const float* __restrict__ w, _Float16* __restrict__ wt,
                        int K, int N) {
  int i = blockIdx.x * blockDim.x + threadIdx.x;
  if (i >= K * N) return;
  int n = i / K;
  int k = i - n * K;
  wt[i] = (_Float16)w[(size_t)k * N + n];
}

// ---------------- segment mean ----------------
// One block per sample; 256 threads own 3 columns each; monotone seg ids ->
// uniform boundary flushes, no atomics. Coalesced 1024B-wide streaming loads.
__global__ __launch_bounds__(256)
void k_segmean(const float* __restrict__ hidden, const int* __restrict__ seg,
               _Float16* __restrict__ sentem) {
  __shared__ int sid[Lq];
  const int b   = blockIdx.x;
  const int tid = threadIdx.x;
  sid[tid]       = seg[b * Lq + tid];
  sid[tid + 256] = seg[b * Lq + tid + 256];
  __syncthreads();

  const float* hb = hidden + (size_t)b * Lq * Dq;
  _Float16*    ob = sentem + (size_t)b * Sq * Dq;

  float a0 = 0.f, a1 = 0.f, a2 = 0.f;
  int cur = sid[0];
  int cnt = 0;
  for (int t = 0; t < Lq; ++t) {
    int s = sid[t];              // uniform across the block -> scalar branch
    if (s != cur) {
      float inv = 1.0f / (float)cnt;
      ob[cur * Dq + tid]       = (_Float16)(a0 * inv);
      ob[cur * Dq + tid + 256] = (_Float16)(a1 * inv);
      ob[cur * Dq + tid + 512] = (_Float16)(a2 * inv);
      a0 = a1 = a2 = 0.f;
      cnt = 0;
      cur = s;
    }
    const float* hp = hb + (size_t)t * Dq;
    a0 += hp[tid];
    a1 += hp[tid + 256];
    a2 += hp[tid + 512];
    ++cnt;
  }
  float inv = 1.0f / (float)cnt;
  ob[cur * Dq + tid]       = (_Float16)(a0 * inv);
  ob[cur * Dq + tid + 256] = (_Float16)(a1 * inv);
  ob[cur * Dq + tid + 512] = (_Float16)(a2 * inv);
}

// ---------------- fused MLP (WMMA) ----------------
// 16 rows per block, 8 waves. Layer1: wave handles 3 of 24 N-tiles, K=24 steps.
// Layer2: wave handles 1 of 8 N-tiles, K=12 steps. Layer3 (N=2): VALU on wave 0.
__global__ __launch_bounds__(256)
void k_mlp(const _Float16* __restrict__ X,    // [4096][768]
           const _Float16* __restrict__ W1t,  // [384][768]  (N-major)
           const float*    __restrict__ b1,   // [384]
           const _Float16* __restrict__ W2t,  // [128][384]  (N-major)
           const float*    __restrict__ b2,   // [128]
           const float*    __restrict__ W3,   // [128][2]
           const float*    __restrict__ b3,   // [2]
           float*          __restrict__ out)  // [4096][2]
{
  __shared__ __attribute__((aligned(16))) _Float16 Xs [16 * XS_STRIDE];
  __shared__ __attribute__((aligned(16))) _Float16 H1s[16 * H1_STRIDE];
  __shared__ __attribute__((aligned(16))) _Float16 H2s[16 * H2_STRIDE];

  const int tid  = threadIdx.x;
  const int wave = tid >> 5;
  const int lane = tid & 31;
  const int m0   = blockIdx.x * 16;

  // Cooperative load of the 16x768 X tile (contiguous 24KB in global).
  {
    const uint4* gx = (const uint4*)(X + (size_t)m0 * Dq); // 16 rows x 96 uint4
    uint4* lx = (uint4*)Xs;                                // row stride 97 uint4
    for (int i = tid; i < 16 * 96; i += 256) {
      int r = i / 96, p = i - r * 96;
      lx[r * 97 + p] = gx[i];
    }
  }
  __syncthreads();

  const int mrow  = lane & 15;
  const int hi    = lane >> 4;    // 0 or 1 (lane half)
  const int abase = hi * 8;       // A frag: K sub-base (halfs)
  const int bbase = hi * 8;       // B frag: K sub-base (halfs)
  const int ncolL = lane & 15;

  // ---------- Layer 1: [16x768] x [768x384] ----------
  v8f acc[3] = {};
  for (int kk = 0; kk < 24; ++kk) {
    const _Float16* ap = &Xs[mrow * XS_STRIDE + kk * 32 + abase];
    v8h alo = *(const v8h*)(ap);
    v8h ahi = *(const v8h*)(ap + 16);
    v16h a;
#pragma unroll
    for (int i = 0; i < 8; ++i) { a[i] = alo[i]; a[i + 8] = ahi[i]; }
#pragma unroll
    for (int j = 0; j < 3; ++j) {
      const int ntile = wave + j * 8;
      const _Float16* bp = &W1t[(size_t)(ntile * 16 + ncolL) * Dq + kk * 32 + bbase];
      v8h blo = *(const v8h*)(bp);
      v8h bhi = *(const v8h*)(bp + 16);
      v16h bm;
#pragma unroll
      for (int i = 0; i < 8; ++i) { bm[i] = blo[i]; bm[i + 8] = bhi[i]; }
      acc[j] = __builtin_amdgcn_wmma_f32_16x16x32_f16(
          false, a, false, bm, (short)0, acc[j], false, false);
    }
  }
#pragma unroll
  for (int j = 0; j < 3; ++j) {
    const int ncol = (wave + j * 8) * 16 + ncolL;
    const float bb = b1[ncol];
#pragma unroll
    for (int r = 0; r < 8; ++r) {
      float h = acc[j][r] + bb;
      h = h > 0.f ? h : 0.f;
      H1s[(r + hi * 8) * H1_STRIDE + ncol] = (_Float16)h;
    }
  }
  __syncthreads();

  // ---------- Layer 2: [16x384] x [384x128] ----------
  v8f acc2 = {};
  for (int kk = 0; kk < 12; ++kk) {
    const _Float16* ap = &H1s[mrow * H1_STRIDE + kk * 32 + abase];
    v8h alo = *(const v8h*)(ap);
    v8h ahi = *(const v8h*)(ap + 16);
    v16h a;
#pragma unroll
    for (int i = 0; i < 8; ++i) { a[i] = alo[i]; a[i + 8] = ahi[i]; }
    const _Float16* bp = &W2t[(size_t)(wave * 16 + ncolL) * F1 + kk * 32 + bbase];
    v8h blo = *(const v8h*)(bp);
    v8h bhi = *(const v8h*)(bp + 16);
    v16h bm;
#pragma unroll
    for (int i = 0; i < 8; ++i) { bm[i] = blo[i]; bm[i + 8] = bhi[i]; }
    acc2 = __builtin_amdgcn_wmma_f32_16x16x32_f16(
        false, a, false, bm, (short)0, acc2, false, false);
  }
  {
    const int ncol = wave * 16 + ncolL;
    const float bb = b2[ncol];
#pragma unroll
    for (int r = 0; r < 8; ++r) {
      float h = acc2[r] + bb;
      h = h > 0.f ? h : 0.f;
      H2s[(r + hi * 8) * H2_STRIDE + ncol] = (_Float16)h;
    }
  }
  __syncthreads();

  // ---------- Layer 3: [16x128] x [128x2] (VALU) ----------
  if (wave == 0) {
    const int m = lane >> 1;
    const int c = lane & 1;
    float s = b3[c];
    for (int k = 0; k < F2; ++k)
      s += (float)H2s[m * H2_STRIDE + k] * W3[k * 2 + c];
    out[(size_t)(m0 + m) * 2 + c] = s;
  }
}

// ---------------- launcher ----------------

extern "C" void kernel_launch(void* const* d_in, const int* in_sizes, int n_in,
                              void* d_out, int out_size, void* d_ws, size_t ws_size,
                              hipStream_t stream) {
  const float* hidden = (const float*)d_in[0];
  const int*   seg    = (const int*)d_in[1];
  const float* W1     = (const float*)d_in[2];
  const float* b1     = (const float*)d_in[3];
  const float* W2     = (const float*)d_in[4];
  const float* b2     = (const float*)d_in[5];
  const float* W3     = (const float*)d_in[6];
  const float* b3     = (const float*)d_in[7];
  float* out = (float*)d_out;

  // Workspace layout (f16): sent_em [4096][768], W1t [384][768], W2t [128][384]
  _Float16* sentem = (_Float16*)d_ws;
  _Float16* W1t    = sentem + (size_t)M_TOTAL * Dq;   // + 3,145,728 halfs
  _Float16* W2t    = W1t    + (size_t)Dq * F1;        // +   294,912 halfs

  // Zero sent_em (empty segments must stay 0)
  {
    int n16 = (M_TOTAL * Dq * 2) / 16;                // 393,216 uint4
    k_zero16<<<(n16 + 255) / 256, 256, 0, stream>>>((uint4*)sentem, n16);
  }
  // Convert + transpose weights to f16 N-major
  k_cvt_t<<<(Dq * F1 + 255) / 256, 256, 0, stream>>>(W1, W1t, Dq, F1);
  k_cvt_t<<<(F1 * F2 + 255) / 256, 256, 0, stream>>>(W2, W2t, F1, F2);

  // Segment mean (memory-bound phase: streams 100.7 MB of hidden)
  k_segmean<<<Bq, 256, 0, stream>>>(hidden, seg, sentem);

  // Fused WMMA MLP
  k_mlp<<<M_TOTAL / 16, 256, 0, stream>>>(sentem, W1t, b1, W2t, b2, W3, b3, out);
}
